// DOC_RNN_67035849556552
// MI455X (gfx1250) — compile-verified, matrix operand order checked by reading, not run on hardware
//
#include <hip/hip_runtime.h>

#define B_   32
#define T_   2048
#define E_   256
#define H_   256
#define LBL_ 20

typedef __bf16 bf16x8 __attribute__((ext_vector_type(8)));
typedef __bf16 v16bf  __attribute__((ext_vector_type(16)));
typedef float  v8f    __attribute__((ext_vector_type(8)));

// ---------------- helpers ----------------

__device__ __forceinline__ unsigned short f2bf(float f) {
    unsigned u = __float_as_uint(f);
    unsigned r = u + 0x7FFFu + ((u >> 16) & 1u);   // round-to-nearest-even
    return (unsigned short)(r >> 16);
}

// Hardware tanh (gfx1250 V_TANH_F32). Builtin when available, else inline asm
// with trailing v_nops per the TRANS-op co-execution hazard rule.
__device__ __forceinline__ float fast_tanh(float x) {
#if __has_builtin(__builtin_amdgcn_tanhf)
    return __builtin_amdgcn_tanhf(x);
#elif __has_builtin(__builtin_amdgcn_tanh_f32)
    return __builtin_amdgcn_tanh_f32(x);
#else
    float y;
    asm("v_tanh_f32_e32 %0, %1\n\tv_nop\n\tv_nop" : "=v"(y) : "v"(x));
    return y;
#endif
}

__device__ __forceinline__ float fast_sigmoid(float x) {
    return __builtin_fmaf(0.5f, fast_tanh(0.5f * x), 0.5f);
}

// B-fragment (weights, row-major [N,K]) lane slice: two 16B chunks.
__device__ __forceinline__ v16bf load_wfrag(const unsigned short* rowbase, int e0) {
    union { v16bf v; bf16x8 h[2]; } u;
    u.h[0] = *reinterpret_cast<const bf16x8*>(rowbase + e0);
    u.h[1] = *reinterpret_cast<const bf16x8*>(rowbase + e0 + 16);
    return u.v;
}

// ---------------- kernels ----------------

// plain f32 -> bf16 convert (weights)
__global__ void k_cvt_bf16(const float* __restrict__ src,
                           unsigned short* __restrict__ dst, int n4) {
    int i = blockIdx.x * blockDim.x + threadIdx.x;
    if (i < n4) {
        float4 f = reinterpret_cast<const float4*>(src)[i];
        ushort4 o;
        o.x = f2bf(f.x); o.y = f2bf(f.y); o.z = f2bf(f.z); o.w = f2bf(f.w);
        reinterpret_cast<ushort4*>(dst)[i] = o;
    }
}

// f32 -> bf16 convert of X with swizzle into WMMA A-fragment tile order:
// Xsw[(b*128+tt)*4096 + kk*512 + lane*16 + c*8 + j]
// where t = tt*16+m, e = kk*32 + half*8 + c*16 + j, lane = half*16+m.
// Each lane's per-k-step 32B fragment slice becomes contiguous & coalesced.
__global__ void k_cvt_swizzle_x(const float* __restrict__ X,
                                unsigned short* __restrict__ Xsw, int ngroups) {
    int g = blockIdx.x * blockDim.x + threadIdx.x;   // one 8-element group
    if (g >= ngroups) return;
    const int e0 = (g & 31) << 3;              // 32 groups per 256-wide row
    const int t  = (g >> 5) & (T_ - 1);
    const int b  = g >> 16;                    // 2^16 groups per batch
    const float* src = X + ((size_t)(b * T_ + t)) * E_ + e0;
    float4 f0 = reinterpret_cast<const float4*>(src)[0];
    float4 f1 = reinterpret_cast<const float4*>(src)[1];
    const int kk   = e0 >> 5;
    const int r    = e0 & 31;
    const int half = (r >> 3) & 1;
    const int c    = (r >> 4) & 1;
    const int m    = t & 15;
    const int tt   = t >> 4;
    const int lane = half * 16 + m;
    size_t doff = ((size_t)(b * 128 + tt)) * 4096 + kk * 512 + lane * 16 + c * 8;
    uint4 w;
    w.x = (unsigned)f2bf(f0.x) | ((unsigned)f2bf(f0.y) << 16);
    w.y = (unsigned)f2bf(f0.z) | ((unsigned)f2bf(f0.w) << 16);
    w.z = (unsigned)f2bf(f1.x) | ((unsigned)f2bf(f1.y) << 16);
    w.w = (unsigned)f2bf(f1.z) | ((unsigned)f2bf(f1.w) << 16);
    *reinterpret_cast<uint4*>(Xsw + doff) = w;
}

// Fused: gates GEMM (i/g/o only; f-gate dead since c0==0) -> LSTM nonlinearity
// (hardware v_tanh_f32) -> masked avg+max pool. One wave per (b, 16 channels).
// t-loop bounded at ceil(L/16)*16: fully-masked tiles are skipped entirely.
__global__ __launch_bounds__(32)
void k_lstm_pool(const unsigned short* __restrict__ Xsw,   // swizzled bf16 X
                 const unsigned short* __restrict__ Wfb,   // [4H,E] bf16
                 const unsigned short* __restrict__ Wrb,   // [4H,E] bf16
                 const float* __restrict__ b_ih_f, const float* __restrict__ b_hh_f,
                 const float* __restrict__ b_ih_r, const float* __restrict__ b_hh_r,
                 const int*   __restrict__ lengths,
                 float*       __restrict__ doc)            // [B, 4H] (avg | max)
{
    const int wid = blockIdx.x;          // 0..B*32-1
    const int b   = wid >> 5;
    const int c0  = (wid & 31) << 4;     // channel base in [0, 2H)
    const int dir = c0 >> 8;             // 0 = fwd, 1 = rev
    const int hid = c0 & (H_ - 1);

    const unsigned short* W  = dir ? Wrb : Wfb;
    const float* bih = dir ? b_ih_r : b_ih_f;
    const float* bhh = dir ? b_hh_r : b_hh_f;

    const int l    = threadIdx.x;        // lane 0..31
    const int n    = l & 15;             // column within tile
    const int ksel = (l >> 4) << 3;      // K sub-select for W fragments
    const int rofs = (l >> 4) << 3;      // C-layout row offset (0 or 8)

    // gate rows in W (torch order i,f,g,o); f skipped
    const int ri = hid + n;
    const int rg = 2 * H_ + hid + n;
    const int ro = 3 * H_ + hid + n;

    const float bi = bih[ri] + bhh[ri];
    const float bg = bih[rg] + bhh[rg];
    const float bo = bih[ro] + bhh[ro];

    // Hoist all B-matrix fragments (loop-invariant over T): 24 x 8 VGPRs
    v16bf Bi[8], Bg[8], Bo[8];
#pragma unroll
    for (int kk = 0; kk < 8; ++kk) {
        const int e0 = kk * 32 + ksel;
        Bi[kk] = load_wfrag(W + (size_t)ri * E_, e0);
        Bg[kk] = load_wfrag(W + (size_t)rg * E_, e0);
        Bo[kk] = load_wfrag(W + (size_t)ro * E_, e0);
    }

    int L = lengths[b];
    L = L < 1 ? 1 : (L > T_ ? T_ : L);
    const int Tlim = (L + 15) & ~15;     // skip fully-masked tiles

    const unsigned short* Xb = Xsw + (size_t)b * T_ * E_;
    const int lo = l * 16;               // lane offset within a k-step block

    float s_sum = 0.0f;
    float s_max = -1e30f;

    // software-pipelined A fragment (double buffer + next-tile prefetch)
    v16bf a = *reinterpret_cast<const v16bf*>(Xb + lo);

#pragma unroll 1
    for (int t0 = 0; t0 < Tlim; t0 += 16) {
        const unsigned short* Xt = Xb + (size_t)(t0 >> 4) * 4096;
        const unsigned short* Xn = Xb + (size_t)(((t0 + 16) & (T_ - 1)) >> 4) * 4096;
        v8f ai = {0,0,0,0,0,0,0,0};
        v8f ag = {0,0,0,0,0,0,0,0};
        v8f ao = {0,0,0,0,0,0,0,0};
#pragma unroll
        for (int kk = 0; kk < 8; ++kk) {
            v16bf an = (kk < 7)
                ? *reinterpret_cast<const v16bf*>(Xt + (kk + 1) * 512 + lo)
                : *reinterpret_cast<const v16bf*>(Xn + lo);
            ai = __builtin_amdgcn_wmma_f32_16x16x32_bf16(false, a, false, Bi[kk],
                                                         (short)0, ai, false, false);
            ag = __builtin_amdgcn_wmma_f32_16x16x32_bf16(false, a, false, Bg[kk],
                                                         (short)0, ag, false, false);
            ao = __builtin_amdgcn_wmma_f32_16x16x32_bf16(false, a, false, Bo[kk],
                                                         (short)0, ao, false, false);
            a = an;
        }
        // activation + masked pooling; C layout: VGPR r -> row (rofs + r)
#pragma unroll
        for (int r = 0; r < 8; ++r) {
            const int   tg = t0 + rofs + r;
            const float gi = ai[r] + bi;
            const float gg = ag[r] + bg;
            const float go = ao[r] + bo;
            const float cc = fast_sigmoid(gi) * fast_tanh(gg);
            const float hh = fast_sigmoid(go) * fast_tanh(cc);
            const bool  in = tg < L;
            s_sum += in ? hh : 0.0f;                  // v_cndmask, no branch
            s_max  = fmaxf(s_max, in ? hh : -1e30f);  // v_cndmask, no branch
        }
    }

    // combine lane halves (rows 0-7 vs 8-15 live in l and l^16)
    s_sum += __shfl_xor(s_sum, 16, 32);
    s_max  = fmaxf(s_max, __shfl_xor(s_max, 16, 32));

    if (l < 16) {
        const int c = c0 + n;                         // channel in [0, 2H)
        doc[(size_t)b * 4 * H_ + c]          = s_sum / (float)L;  // avg pool
        doc[(size_t)b * 4 * H_ + 2 * H_ + c] = s_max;             // max pool
    }
}

// d1[b,j] = b1[j] + sum_k relu(doc[b,k]) * W1[j,k]
__global__ __launch_bounds__(256)
void k_mlp1(const float* __restrict__ doc, const float* __restrict__ W1,
            const float* __restrict__ b1, float* __restrict__ d1) {
    const int b = blockIdx.x;
    const int j = threadIdx.x;
    const float* dr = doc + (size_t)b * 1024;
    const float* wr = W1 + (size_t)j * 1024;
    float acc = b1[j];
    for (int k = 0; k < 1024; ++k) acc += fmaxf(dr[k], 0.0f) * wr[k];
    d1[b * 256 + j] = acc;
}

// out[b,j] = b2[j] + sum_k d1[b,k] * W2[j,k]
__global__ void k_mlp2(const float* __restrict__ d1, const float* __restrict__ W2,
                       const float* __restrict__ b2, float* __restrict__ out) {
    const int idx = blockIdx.x * blockDim.x + threadIdx.x;
    if (idx < B_ * LBL_) {
        const int b = idx / LBL_, j = idx % LBL_;
        const float* dr = d1 + (size_t)b * 256;
        const float* wr = W2 + (size_t)j * 256;
        float acc = b2[j];
        for (int k = 0; k < 256; ++k) acc += dr[k] * wr[k];
        out[idx] = acc;
    }
}

// ---------------- launcher ----------------

extern "C" void kernel_launch(void* const* d_in, const int* in_sizes, int n_in,
                              void* d_out, int out_size, void* d_ws, size_t ws_size,
                              hipStream_t stream) {
    const float* X      = (const float*)d_in[0];
    const int*   len    = (const int*)  d_in[1];
    const float* W_ih_f = (const float*)d_in[2];
    const float* b_ih_f = (const float*)d_in[3];
    const float* b_hh_f = (const float*)d_in[4];
    const float* W_ih_r = (const float*)d_in[5];
    const float* b_ih_r = (const float*)d_in[6];
    const float* b_hh_r = (const float*)d_in[7];
    const float* W1     = (const float*)d_in[8];
    const float* b1     = (const float*)d_in[9];
    const float* W2     = (const float*)d_in[10];
    const float* b2     = (const float*)d_in[11];
    float* out = (float*)d_out;

    // workspace layout (all offsets 256B-aligned)
    const size_t XBF_B = (size_t)B_ * T_ * E_ * 2;       // 33,554,432
    const size_t W_B   = (size_t)4 * H_ * E_ * 2;        //    524,288
    char* ws = (char*)d_ws;
    unsigned short* Xsw = (unsigned short*)ws;
    unsigned short* Wfb = (unsigned short*)(ws + XBF_B);
    unsigned short* Wrb = (unsigned short*)(ws + XBF_B + W_B);
    float* doc = (float*)(ws + XBF_B + 2 * W_B);
    float* d1  = (float*)(ws + XBF_B + 2 * W_B + (size_t)B_ * 4 * H_ * 4);

    // 1) convert: X (with fragment swizzle), W_ih_f, W_ih_r to bf16
    {
        int ng = (B_ * T_ * E_) / 8;   // 8-element groups
        k_cvt_swizzle_x<<<(ng + 255) / 256, 256, 0, stream>>>(X, Xsw, ng);
        int w4 = (4 * H_ * E_) / 4;
        k_cvt_bf16<<<(w4 + 255) / 256, 256, 0, stream>>>(W_ih_f, Wfb, w4);
        k_cvt_bf16<<<(w4 + 255) / 256, 256, 0, stream>>>(W_ih_r, Wrb, w4);
    }

    // 2) fused WMMA gates + LSTM cell + masked avg/max pool -> doc [B, 4H]
    k_lstm_pool<<<B_ * 32, 32, 0, stream>>>(Xsw, Wfb, Wrb,
                                            b_ih_f, b_hh_f, b_ih_r, b_hh_r,
                                            len, doc);

    // 3) MLP head
    k_mlp1<<<B_, 256, 0, stream>>>(doc, W1, b1, d1);
    k_mlp2<<<(B_ * LBL_ + 63) / 64, 64, 0, stream>>>(d1, W2, b2, out);
}
